// OctoTransformer_73323681677340
// MI455X (gfx1250) — matrix operand order repaired and checked by reading, not compiled, not run on hardware
//
#include <hip/hip_runtime.h>
#include <hip/hip_bf16.h>

// ---------------------------------------------------------------- constants
#define B_      8
#define T_      2
#define NPATCH_ 256
#define SDIM_   14
#define IMGD_   512
#define E_      768
#define H_      12
#define DH_     64
#define FFN_    3072
#define NL_     6
#define NT_     257          // 1 + NPATCH
#define NR_     1
#define STEP_   258          // NT_ + NR_
#define L_      516          // STEP_ * T_
#define M_      (B_ * L_)    // 4128 rows in the residual stream

// ---------------------------------------------------------------- types
typedef __bf16 bf16;
typedef __attribute__((ext_vector_type(8)))  bf16  bf16x8;
typedef __attribute__((ext_vector_type(16))) bf16  bf16x16;
typedef __attribute__((ext_vector_type(8)))  float f32x8;

union V16 { bf16x16 v; bf16x8 h[2]; };

static __device__ inline bf16 f2bf(float f) {
    union { float f; unsigned u; } x; x.f = f;
    unsigned r = x.u + 0x7FFFu + ((x.u >> 16) & 1u);   // round-to-nearest-even
    union { unsigned short s; bf16 b; } y; y.s = (unsigned short)(r >> 16);
    return y.b;
}

// ---------------------------------------------------------------- fp32 -> bf16 cast
__global__ void cast_f32_bf16(const float* __restrict__ in, bf16* __restrict__ out, int n) {
    int i = blockIdx.x * blockDim.x + threadIdx.x;
    if (i < n) out[i] = f2bf(in[i]);
}

// ---------------------------------------------------------------- WMMA GEMM
// C[M,N] = A[M,K](bf16) @ W[N,K]^T(bf16) + bias[N]
// One wave computes a 32x64 output tile: 2 A fragments x 4 B fragments ->
// 8 v_wmma_f32_16x16x32_bf16 per 32-wide K-step, software-pipelined so the
// next K-step's operand loads are in flight while the current WMMAs execute.
// MODE 0: Cf = acc + bias              (fp32 out)
// MODE 1: Cf += acc + bias             (residual accumulate into fp32 stream)
// MODE 2: Cb = bf16(gelu(acc + bias))  (exact GELU, bf16 out)
template<int MODE>
__global__ void gemm_wmma_bf16(const bf16* __restrict__ A, const bf16* __restrict__ W,
                               const float* __restrict__ bias,
                               float* __restrict__ Cf, bf16* __restrict__ Cb,
                               int N, int K) {
    const int lane  = threadIdx.x;          // 0..31
    const int mTile = blockIdx.x * 32;
    const int nBase = blockIdx.y * 64;
    const int sub   = lane & 15;            // M row (A) / N col (B) within 16-tile
    const int khA   = (lane >> 4) << 3;     // A K-half offset: 0 or 8
    const int kbB   = (lane >> 4) << 3;     // B K-half offset: 0 or 8

    const bf16* Ar0 = A + (size_t)(mTile + sub) * K + khA;   // M sub-tile 0
    const bf16* Ar1 = Ar0 + (size_t)16 * K;                  // M sub-tile 1
    const bf16* Wr  = W + (size_t)(nBase + sub) * K + kbB;
    const size_t WS = (size_t)16 * K;                        // N sub-tile stride

    f32x8 acc[2][4];
#pragma unroll
    for (int m = 0; m < 2; ++m)
#pragma unroll
        for (int s = 0; s < 4; ++s)
            acc[m][s] = (f32x8){0.f,0.f,0.f,0.f,0.f,0.f,0.f,0.f};

    // ---- preload K-step 0 ----
    V16 a[2], b[4];
    a[0].h[0] = *(const bf16x8*)(Ar0);      a[0].h[1] = *(const bf16x8*)(Ar0 + 16);
    a[1].h[0] = *(const bf16x8*)(Ar1);      a[1].h[1] = *(const bf16x8*)(Ar1 + 16);
#pragma unroll
    for (int s = 0; s < 4; ++s) {
        const bf16* wp = Wr + (size_t)s * WS;
        b[s].h[0] = *(const bf16x8*)(wp);   b[s].h[1] = *(const bf16x8*)(wp + 16);
    }

    // ---- pipelined main loop (K >= 64 for all GEMMs here) ----
    int k = 0;
    for (; k < K - 32; k += 32) {
        V16 na[2], nb[4];
        const int kn = k + 32;
        na[0].h[0] = *(const bf16x8*)(Ar0 + kn);      na[0].h[1] = *(const bf16x8*)(Ar0 + kn + 16);
        na[1].h[0] = *(const bf16x8*)(Ar1 + kn);      na[1].h[1] = *(const bf16x8*)(Ar1 + kn + 16);
#pragma unroll
        for (int s = 0; s < 4; ++s) {
            const bf16* wp = Wr + (size_t)s * WS + kn;
            nb[s].h[0] = *(const bf16x8*)(wp);        nb[s].h[1] = *(const bf16x8*)(wp + 16);
        }
#pragma unroll
        for (int m = 0; m < 2; ++m)
#pragma unroll
            for (int s = 0; s < 4; ++s)
                acc[m][s] = __builtin_amdgcn_wmma_f32_16x16x32_bf16(
                    false, a[m].v, false, b[s].v, (short)0, acc[m][s], false, false);
#pragma unroll
        for (int m = 0; m < 2; ++m) a[m] = na[m];
#pragma unroll
        for (int s = 0; s < 4; ++s) b[s] = nb[s];
    }
    // ---- drain last K-step ----
#pragma unroll
    for (int m = 0; m < 2; ++m)
#pragma unroll
        for (int s = 0; s < 4; ++s)
            acc[m][s] = __builtin_amdgcn_wmma_f32_16x16x32_bf16(
                false, a[m].v, false, b[s].v, (short)0, acc[m][s], false, false);

    // ---- epilogue ----
    const int mOff = (lane >> 4) << 3;      // C/D: lanes 16-31 hold M = 8..15
#pragma unroll
    for (int m = 0; m < 2; ++m) {
#pragma unroll
        for (int s = 0; s < 4; ++s) {
            const int col = nBase + s * 16 + sub;
            const float bs = bias[col];
#pragma unroll
            for (int r = 0; r < 8; ++r) {
                const int row = mTile + m * 16 + mOff + r;
                const size_t idx = (size_t)row * N + col;
                float v = acc[m][s][r] + bs;
                if (MODE == 0) {
                    Cf[idx] = v;
                } else if (MODE == 1) {
                    Cf[idx] += v;
                } else {
                    float g = 0.5f * v * (1.0f + erff(v * 0.70710678118654752f));
                    Cb[idx] = f2bf(g);
                }
            }
        }
    }
}

// ---------------------------------------------------------------- LayerNorm -> bf16
__global__ void ln_bf16_kernel(const float* __restrict__ X, const float* __restrict__ w,
                               const float* __restrict__ b, bf16* __restrict__ out) {
    __shared__ float red[256];
    const int row = blockIdx.x;
    const int tid = threadIdx.x;
    const float* x = X + (size_t)row * E_;
    float v0 = x[tid], v1 = x[tid + 256], v2 = x[tid + 512];
    red[tid] = v0 + v1 + v2;
    __syncthreads();
    for (int o = 128; o > 0; o >>= 1) { if (tid < o) red[tid] += red[tid + o]; __syncthreads(); }
    const float mu = red[0] * (1.0f / E_);
    __syncthreads();
    float d0 = v0 - mu, d1 = v1 - mu, d2 = v2 - mu;
    red[tid] = d0 * d0 + d1 * d1 + d2 * d2;
    __syncthreads();
    for (int o = 128; o > 0; o >>= 1) { if (tid < o) red[tid] += red[tid + o]; __syncthreads(); }
    const float rstd = rsqrtf(red[0] * (1.0f / E_) + 1e-5f);
    bf16* orow = out + (size_t)row * E_;
    orow[tid]       = f2bf(d0 * rstd * w[tid]       + b[tid]);
    orow[tid + 256] = f2bf(d1 * rstd * w[tid + 256] + b[tid + 256]);
    orow[tid + 512] = f2bf(d2 * rstd * w[tid + 512] + b[tid + 512]);
}

// ---------------------------------------------------------------- final LN on readout rows
__global__ void final_ln_kernel(const float* __restrict__ X, const float* __restrict__ w,
                                const float* __restrict__ b, float* __restrict__ out) {
    __shared__ float red[256];
    const int t = blockIdx.x, bb = blockIdx.y;
    const int row = bb * L_ + t * STEP_ + NT_;           // readout token (NR_=1)
    const int tid = threadIdx.x;
    const float* x = X + (size_t)row * E_;
    float v0 = x[tid], v1 = x[tid + 256], v2 = x[tid + 512];
    red[tid] = v0 + v1 + v2;
    __syncthreads();
    for (int o = 128; o > 0; o >>= 1) { if (tid < o) red[tid] += red[tid + o]; __syncthreads(); }
    const float mu = red[0] * (1.0f / E_);
    __syncthreads();
    float d0 = v0 - mu, d1 = v1 - mu, d2 = v2 - mu;
    red[tid] = d0 * d0 + d1 * d1 + d2 * d2;
    __syncthreads();
    for (int o = 128; o > 0; o >>= 1) { if (tid < o) red[tid] += red[tid + o]; __syncthreads(); }
    const float rstd = rsqrtf(red[0] * (1.0f / E_) + 1e-5f);
    float* orow = out + (size_t)(bb * T_ + t) * E_;
    orow[tid]       = d0 * rstd * w[tid]       + b[tid];
    orow[tid + 256] = d1 * rstd * w[tid + 256] + b[tid + 256];
    orow[tid + 512] = d2 * rstd * w[tid + 512] + b[tid + 512];
}

// ---------------------------------------------------------------- token assembly
__global__ void assemble_kernel(const float* __restrict__ state, const float* __restrict__ state_w,
                                const float* __restrict__ state_b, const float* __restrict__ imgp,
                                const float* __restrict__ obs_pos, const float* __restrict__ ro_tok,
                                const float* __restrict__ ro_pos, float* __restrict__ X) {
    const int tok = blockIdx.x, t = blockIdx.y, bb = blockIdx.z;
    const int row = bb * L_ + t * STEP_ + tok;
    float* xr = X + (size_t)row * E_;
    for (int e = threadIdx.x; e < E_; e += 256) {
        float v;
        if (tok == 0) {
            const float* st = state + (size_t)(bb * T_ + t) * SDIM_;
            const float* wr = state_w + (size_t)e * SDIM_;
            float acc = state_b[e];
#pragma unroll
            for (int k = 0; k < SDIM_; ++k) acc += st[k] * wr[k];
            v = acc + obs_pos[((size_t)t * NT_) * E_ + e];
        } else if (tok <= NPATCH_) {
            const int p = tok - 1;
            v = imgp[((size_t)((bb * T_ + t) * NPATCH_ + p)) * E_ + e]
              + obs_pos[((size_t)t * NT_ + tok) * E_ + e];
        } else {
            v = ro_tok[(size_t)t * E_ + e] + ro_pos[(size_t)t * E_ + e];
        }
        xr[e] = v;
    }
}

// ---------------------------------------------------------------- attention
// One block per (query, head, batch). Blockwise-causal mask computed analytically.
__global__ void attn_kernel(const float* __restrict__ QKV, bf16* __restrict__ Aout) {
    __shared__ float qs[DH_];
    __shared__ float sc[L_];
    __shared__ float red[256];
    __shared__ float part[4][DH_];
    const int qi = blockIdx.x, h = blockIdx.y, bb = blockIdx.z;
    const int tid = threadIdx.x;

    if (tid < DH_)
        qs[tid] = QKV[((size_t)(bb * L_ + qi)) * (3 * E_) + h * DH_ + tid] * 0.125f; // 1/sqrt(64)
    __syncthreads();

    for (int j = tid; j < L_; j += 256) {
        // key j visible iff j<=qi and (j is not a readout column unless j==qi)
        const bool ok = (j <= qi) && ((j % STEP_) < NT_ || j == qi);
        float s = -1e30f;
        if (ok) {
            const float* kr = QKV + ((size_t)(bb * L_ + j)) * (3 * E_) + E_ + h * DH_;
            float acc = 0.f;
#pragma unroll
            for (int d = 0; d < DH_; ++d) acc += qs[d] * kr[d];
            s = acc;
        }
        sc[j] = s;
    }
    __syncthreads();

    float lm = -1e30f;
    for (int j = tid; j < L_; j += 256) lm = fmaxf(lm, sc[j]);
    red[tid] = lm;
    __syncthreads();
    for (int o = 128; o > 0; o >>= 1) { if (tid < o) red[tid] = fmaxf(red[tid], red[tid + o]); __syncthreads(); }
    const float mx = red[0];
    __syncthreads();

    float ls = 0.f;
    for (int j = tid; j < L_; j += 256) { float e = expf(sc[j] - mx); sc[j] = e; ls += e; }
    red[tid] = ls;
    __syncthreads();
    for (int o = 128; o > 0; o >>= 1) { if (tid < o) red[tid] += red[tid + o]; __syncthreads(); }
    const float inv = 1.0f / red[0];
    __syncthreads();
    for (int j = tid; j < L_; j += 256) sc[j] *= inv;
    __syncthreads();

    const int d = tid & 63, cj = tid >> 6;
    float acc = 0.f;
    for (int j = cj; j < L_; j += 4)
        acc += sc[j] * QKV[((size_t)(bb * L_ + j)) * (3 * E_) + 2 * E_ + h * DH_ + d];
    part[cj][d] = acc;
    __syncthreads();
    if (cj == 0) {
        float o = part[0][d] + part[1][d] + part[2][d] + part[3][d];
        Aout[((size_t)(bb * L_ + qi)) * E_ + h * DH_ + d] = f2bf(o);
    }
}

// ---------------------------------------------------------------- workspace layout (bytes)
#define OFF_X   ((size_t)0)                                   // fp32 residual  [M_,E_]
#define OFF_H   (OFF_X   + (size_t)M_ * E_ * 4)               // bf16 LN out    [M_,E_]
#define OFF_QKV (OFF_H   + (size_t)M_ * E_ * 2)               // fp32 qkv       [M_,3E_] (aliases img_p)
#define OFF_A   (OFF_QKV + (size_t)M_ * 3 * E_ * 4)           // bf16 attn out  [M_,E_]
#define OFF_G   (OFF_A   + (size_t)M_ * E_ * 2)               // bf16 gelu out  [M_,FFN_]
#define OFF_W   (OFF_G   + (size_t)M_ * FFN_ * 2)             // bf16 weight staging (max FFN_*E_)
#define OFF_I   (OFF_W   + (size_t)FFN_ * E_ * 2)             // bf16 img feats [B_*T_*NPATCH_,IMGD_]

extern "C" void kernel_launch(void* const* d_in, const int* in_sizes, int n_in,
                              void* d_out, int out_size, void* d_ws, size_t ws_size,
                              hipStream_t stream) {
    const float* state   = (const float*)d_in[0];
    const float* imgf    = (const float*)d_in[1];
    const float* state_w = (const float*)d_in[2];
    const float* state_b = (const float*)d_in[3];
    const float* img_w   = (const float*)d_in[4];
    const float* img_b   = (const float*)d_in[5];
    const float* ro_tok  = (const float*)d_in[6];
    const float* obs_pos = (const float*)d_in[7];
    const float* ro_pos  = (const float*)d_in[8];
    const float* qkv_w   = (const float*)d_in[9];
    const float* qkv_b   = (const float*)d_in[10];
    const float* out_w   = (const float*)d_in[11];
    const float* out_b   = (const float*)d_in[12];
    const float* ln1_w   = (const float*)d_in[13];
    const float* ln1_b   = (const float*)d_in[14];
    const float* ln2_w   = (const float*)d_in[15];
    const float* ln2_b   = (const float*)d_in[16];
    const float* ffn_w1  = (const float*)d_in[17];
    const float* ffn_b1  = (const float*)d_in[18];
    const float* ffn_w2  = (const float*)d_in[19];
    const float* ffn_b2  = (const float*)d_in[20];
    const float* lnf_w   = (const float*)d_in[21];
    const float* lnf_b   = (const float*)d_in[22];

    char* ws = (char*)d_ws;
    float* X    = (float*)(ws + OFF_X);
    bf16*  H16  = (bf16*) (ws + OFF_H);
    float* QKV  = (float*)(ws + OFF_QKV);
    float* IMGP = QKV;                       // alias: img projection used before layer 0
    bf16*  A16  = (bf16*) (ws + OFF_A);
    bf16*  G16  = (bf16*) (ws + OFF_G);
    bf16*  WB   = (bf16*) (ws + OFF_W);
    bf16*  I16  = (bf16*) (ws + OFF_I);

    auto cast = [&](const float* src, bf16* dst, int n) {
        cast_f32_bf16<<<dim3((n + 255) / 256), dim3(256), 0, stream>>>(src, dst, n);
    };

    // ---- embeddings -------------------------------------------------------
    cast(imgf,  I16, B_ * T_ * NPATCH_ * IMGD_);
    cast(img_w, WB,  E_ * IMGD_);
    gemm_wmma_bf16<0><<<dim3((B_ * T_ * NPATCH_) / 32, E_ / 64), dim3(32), 0, stream>>>(
        I16, WB, img_b, IMGP, nullptr, E_, IMGD_);
    assemble_kernel<<<dim3(STEP_, T_, B_), dim3(256), 0, stream>>>(
        state, state_w, state_b, IMGP, obs_pos, ro_tok, ro_pos, X);

    // ---- transformer layers ----------------------------------------------
    for (int i = 0; i < NL_; ++i) {
        ln_bf16_kernel<<<dim3(M_), dim3(256), 0, stream>>>(X, ln1_w + i * E_, ln1_b + i * E_, H16);

        cast(qkv_w + (size_t)i * 3 * E_ * E_, WB, 3 * E_ * E_);
        gemm_wmma_bf16<0><<<dim3(M_ / 32, (3 * E_) / 64), dim3(32), 0, stream>>>(
            H16, WB, qkv_b + (size_t)i * 3 * E_, QKV, nullptr, 3 * E_, E_);

        attn_kernel<<<dim3(L_, H_, B_), dim3(256), 0, stream>>>(QKV, A16);

        cast(out_w + (size_t)i * E_ * E_, WB, E_ * E_);
        gemm_wmma_bf16<1><<<dim3(M_ / 32, E_ / 64), dim3(32), 0, stream>>>(
            A16, WB, out_b + (size_t)i * E_, X, nullptr, E_, E_);

        ln_bf16_kernel<<<dim3(M_), dim3(256), 0, stream>>>(X, ln2_w + i * E_, ln2_b + i * E_, H16);

        cast(ffn_w1 + (size_t)i * FFN_ * E_, WB, FFN_ * E_);
        gemm_wmma_bf16<2><<<dim3(M_ / 32, FFN_ / 64), dim3(32), 0, stream>>>(
            H16, WB, ffn_b1 + (size_t)i * FFN_, nullptr, G16, FFN_, E_);

        cast(ffn_w2 + (size_t)i * E_ * FFN_, WB, E_ * FFN_);
        gemm_wmma_bf16<1><<<dim3(M_ / 32, E_ / 64), dim3(32), 0, stream>>>(
            G16, WB, ffn_b2 + (size_t)i * E_, X, nullptr, E_, FFN_);
    }

    // ---- final LN on readout tokens --------------------------------------
    final_ln_kernel<<<dim3(T_, B_), dim3(256), 0, stream>>>(X, lnf_w, lnf_b, (float*)d_out);
}
